// CCAttention_64115271794854
// MI455X (gfx1250) — matrix operand order repaired and checked by reading, not compiled, not run on hardware
//
#include <hip/hip_runtime.h>
#include <hip/hip_bf16.h>

// ---------- types & helpers ----------
typedef __attribute__((ext_vector_type(16))) __bf16 v16bf;
typedef __attribute__((ext_vector_type(8)))  float  v8f;
typedef unsigned short u16;
typedef unsigned int   u32;

__device__ __forceinline__ u16 f2bf(float f) {
    u32 u = __builtin_bit_cast(u32, f);
    u32 r = (u + 0x7FFFu + ((u >> 16) & 1u)) >> 16;   // RNE
    return (u16)r;
}
__device__ __forceinline__ float bf2f(u16 v) {
    return __builtin_bit_cast(float, (u32)v << 16);
}

__device__ __forceinline__ v8f wmma_bf16(v16bf a, v16bf b, v8f c) {
    return __builtin_amdgcn_wmma_f32_16x16x32_bf16(false, a, false, b, (short)0, c, false, false);
}

// fast softplus; Q/K are rounded to bf16 right after, so v_exp/v_log accuracy suffices
__device__ __forceinline__ float softplus_fast(float z) {
    return fmaxf(z, 0.f) + __logf(1.f + __expf(-fabsf(z)));
}

// A fragment: 16x32 (MxK) bf16. lane m = lane&15, half = lane>>4.
// elements: k = kstep*32 + half*8 + {0..7} and +16..  -> two 16B chunks.
__device__ __forceinline__ v16bf load_frag_A(const u16* rowbase, int strideU16, int kstep, int lane) {
    int m = lane & 15, hf = lane >> 4;
    const u16* r = rowbase + m * strideU16 + kstep * 32 + hf * 8;
    union { v16bf v; uint4 q[2]; } u;
    u.q[0] = *(const uint4*)(r);
    u.q[1] = *(const uint4*)(r + 16);
    return u.v;
}
// B fragment: 32x16 (KxN) bf16 from Bs stored as [n][k].
// lane n = lane&15, khalf = lane>>4; k = kstep*32 + khalf*8 + {0..7} and +16.. -> two 16B chunks.
__device__ __forceinline__ v16bf load_frag_B(const u16* rowbase, int strideU16, int kstep, int lane) {
    int n = lane & 15, kh = lane >> 4;
    const u16* r = rowbase + n * strideU16 + kstep * 32 + kh * 8;
    union { v16bf v; uint4 q[2]; } u;
    u.q[0] = *(const uint4*)(r);
    u.q[1] = *(const uint4*)(r + 16);
    return u.v;
}

#define NEGMASK (-1.0e4f)

// ---------- kernel 1: fused QKV projection (WMMA GEMM [80x64]x[64x256]) ----------
__global__ __launch_bounds__(256) void proj_kernel(
    const float* __restrict__ x,
    const float* __restrict__ wq, const float* __restrict__ bq,
    const float* __restrict__ wk, const float* __restrict__ bk,
    const float* __restrict__ wvw, const float* __restrict__ bv,
    u16* __restrict__ Qb, u16* __restrict__ Kb, u16* __restrict__ Vb)
{
    __shared__ __align__(16) unsigned char sm[48704];
    u16* As   = (u16*)sm;             // [80][72] bf16 weights (rows: 8 q, 8 k, 64 v)
    u16* Bs   = (u16*)(sm + 11520);   // [256 px][72] bf16 x-tile (transposed)
    float* bS = (float*)(sm + 48384); // [80] biases

    const int tid = threadIdx.x;
    const int b   = blockIdx.x >> 6;
    const int P0  = (blockIdx.x & 63) << 8;    // pixel base in plane

    for (int e = tid; e < 5120; e += 256) {
        int o = e >> 6, c = e & 63;
        float v = (o < 8) ? wq[o * 64 + c] : (o < 16) ? wk[(o - 8) * 64 + c] : wvw[(o - 16) * 64 + c];
        As[o * 72 + c] = f2bf(v);
    }
    if (tid < 80) bS[tid] = (tid < 8) ? bq[tid] : (tid < 16) ? bk[tid - 8] : bv[tid - 16];

    const float* xb = x + (size_t)b * 64 * 16384 + P0;
    for (int e = tid; e < 16384; e += 256) {
        int c = e >> 8, pp = e & 255;
        Bs[pp * 72 + c] = f2bf(xb[(size_t)c * 16384 + pp]);
    }
    __syncthreads();

    const int wid = tid >> 5, lane = tid & 31;
    const int hf = lane >> 4, ln = lane & 15;
    for (int i = 0; i < 10; ++i) {
        int t  = wid * 10 + i;
        int mt = t >> 4, nt = t & 15;            // uniform within the wave
        v16bf a0 = load_frag_A(As + mt * 16 * 72, 72, 0, lane);
        v16bf a1 = load_frag_A(As + mt * 16 * 72, 72, 1, lane);
        v16bf b0 = load_frag_B(Bs + nt * 16 * 72, 72, 0, lane);
        v16bf b1 = load_frag_B(Bs + nt * 16 * 72, 72, 1, lane);
        v8f acc = {};
        acc = wmma_bf16(a0, b0, acc);
        acc = wmma_bf16(a1, b1, acc);
        int pp = nt * 16 + ln;
        size_t gp = (size_t)P0 + pp;
        if (mt == 0) {
            // rows 0..15: lanes 0-15 -> Q row r, lanes 16-31 -> K row r (pointer select, no branch)
            u16* dst = (hf ? Kb : Qb) + (size_t)b * 131072 + gp;
#pragma unroll
            for (int r = 0; r < 8; ++r) {
                float v = acc[r] + bS[r + hf * 8];
                dst[(size_t)r * 16384] = f2bf(softplus_fast(v));
            }
        } else {
            // rows 16..79: V channels (mt uniform -> uniform branch)
            u16* dst = Vb + (size_t)b * 1048576 + (size_t)(mt * 16 - 16 + hf * 8) * 16384 + gp;
#pragma unroll
            for (int r = 0; r < 8; ++r) {
                float v = acc[r] + bS[mt * 16 + r + hf * 8];
                dst[(size_t)r * 16384] = f2bf(v);
            }
        }
    }
}

// ---------- kernel 2: 128x128 bf16 plane transpose (for column-pass layouts) ----------
__global__ __launch_bounds__(256) void transpose_kernel(
    const u16* __restrict__ src, u16* __restrict__ dst)
{
    __shared__ u16 Ts[128 * 136];
    const size_t base = (size_t)blockIdx.x * 16384;
    const int tid = threadIdx.x;
    for (int e = tid; e < 16384; e += 256) {
        int i = e >> 7, j = e & 127;
        Ts[i * 136 + j] = src[base + e];
    }
    __syncthreads();
    for (int e = tid; e < 16384; e += 256) {
        int j = e >> 7, i = e & 127;
        dst[base + e] = Ts[i * 136 + j];
    }
}

// ---------- kernel 3: fused attention along one line (shared by column & row pass) ----------
// block = (b, line); E = Q^T K (128x128, K=8 padded to 32), O = V * E^T (64x128).
// MASK pass adds NEG*V[c,line_pos] (the -1e4 diagonal) exactly, post-matmul.
template <bool MASK>
__global__ __launch_bounds__(256) void attn_kernel(
    const u16* __restrict__ Qb, const u16* __restrict__ Kb,
    const u16* __restrict__ Vb, float* __restrict__ Out,
    const float* __restrict__ gammaP)
{
    __shared__ __align__(16) unsigned char sm[52224];
    u16* Qs = (u16*)sm;             // [128][40]  (aliased by Es after barrier)
    u16* Ks = (u16*)(sm + 10240);   // [128][40]
    u16* Es = (u16*)sm;             // [128][136] bf16 energies
    u16* Vs = (u16*)(sm + 34816);   // [64][136]

    const int tid  = threadIdx.x;
    const int b    = blockIdx.x >> 7;
    const int line = blockIdx.x & 127;
    const size_t qkBase = (size_t)b * 131072  + (size_t)line * 128;
    const size_t vBase  = (size_t)b * 1048576 + (size_t)line * 128;

    for (int e = tid; e < 1024; e += 256) {
        int qc = e >> 7, pos = e & 127;
        Qs[pos * 40 + qc] = Qb[qkBase + (size_t)qc * 16384 + pos];
        Ks[pos * 40 + qc] = Kb[qkBase + (size_t)qc * 16384 + pos];
    }
    for (int e = tid; e < 3072; e += 256) {      // zero-pad K-dim 8..31
        int qc = 8 + (e >> 7), pos = e & 127;
        Qs[pos * 40 + qc] = 0;
        Ks[pos * 40 + qc] = 0;
    }
    for (int e = tid; e < 8192; e += 256) {
        int c = e >> 7, pos = e & 127;
        Vs[c * 136 + pos] = Vb[vBase + (size_t)c * 16384 + pos];
    }
    __syncthreads();

    const int wid = tid >> 5, lane = tid & 31;
    const int hf = lane >> 4, ln = lane & 15;

    // matmul1: wave wid owns E rows [wid*16, wid*16+16); keep tiles in regs
    v8f etile[8];
    {
        v16bf a = load_frag_A(Qs + wid * 16 * 40, 40, 0, lane);
#pragma unroll
        for (int nt = 0; nt < 8; ++nt) {
            v16bf bm = load_frag_B(Ks + nt * 16 * 40, 40, 0, lane);
            v8f z = {};
            etile[nt] = wmma_bf16(a, bm, z);
        }
    }
    __syncthreads();   // Qs/Ks dead -> Es may overwrite
#pragma unroll
    for (int nt = 0; nt < 8; ++nt) {
#pragma unroll
        for (int r = 0; r < 8; ++r) {
            int m = wid * 16 + r + hf * 8;
            int n = nt * 16 + ln;
            Es[m * 136 + n] = f2bf(etile[nt][r]);
        }
    }
    __syncthreads();

    const float scale = gammaP ? gammaP[0] : 1.0f;

    // matmul2: wave -> ct = wid>>1, ht in [(wid&1)*4, +4); K = 128 (4 steps)
    const int ct  = wid >> 1;
    const int ht0 = (wid & 1) * 4;
    v16bf afr[4];
#pragma unroll
    for (int ks = 0; ks < 4; ++ks)
        afr[ks] = load_frag_A(Vs + ct * 16 * 136, 136, ks, lane);
#pragma unroll
    for (int i = 0; i < 4; ++i) {
        int ht = ht0 + i;
        v8f acc = {};
#pragma unroll
        for (int ks = 0; ks < 4; ++ks) {
            v16bf bm = load_frag_B(Es + ht * 16 * 136, 136, ks, lane);
            acc = wmma_bf16(afr[ks], bm, acc);
        }
        int y = ht * 16 + ln;
#pragma unroll
        for (int r = 0; r < 8; ++r) {
            int c = ct * 16 + r + hf * 8;
            float v = acc[r];
            if (MASK) v += NEGMASK * bf2f(Vs[c * 136 + y]);   // exact diagonal term
            Out[vBase + (size_t)c * 16384 + y] = scale * v;
        }
    }
}

// ---------- kernel 4: out = d_out(=gamma*oW) + gamma*oH^T + x ----------
__global__ __launch_bounds__(256) void combine_kernel(
    const float* __restrict__ oHt, const float* __restrict__ x,
    const float* __restrict__ gammaP, float* __restrict__ out)
{
    __shared__ float Ts[64 * 137];
    const int tid   = threadIdx.x;
    const int plane = blockIdx.x >> 1;          // b*64 + c
    const int w0    = (blockIdx.x & 1) * 64;
    const size_t pbase = (size_t)plane * 16384;
    const float g = gammaP[0];

    for (int e = tid; e < 8192; e += 256) {
        int wl = e >> 7, h = e & 127;
        Ts[wl * 137 + h] = oHt[pbase + (size_t)(w0 + wl) * 128 + h];
    }
    __syncthreads();
    for (int e = tid; e < 8192; e += 256) {
        int h = e >> 6, wl = e & 63;
        size_t idx = pbase + (size_t)h * 128 + (w0 + wl);
        out[idx] = out[idx] + g * Ts[wl * 137 + h] + x[idx];
    }
}

// ---------- host ----------
extern "C" void kernel_launch(void* const* d_in, const int* in_sizes, int n_in,
                              void* d_out, int out_size, void* d_ws, size_t ws_size,
                              hipStream_t stream) {
    (void)in_sizes; (void)n_in; (void)out_size; (void)ws_size;
    const float* x     = (const float*)d_in[0];
    const float* wq    = (const float*)d_in[1];
    const float* bq    = (const float*)d_in[2];
    const float* wk    = (const float*)d_in[3];
    const float* bk    = (const float*)d_in[4];
    const float* wvw   = (const float*)d_in[5];
    const float* bv    = (const float*)d_in[6];
    const float* gamma = (const float*)d_in[7];
    float* out = (float*)d_out;

    char* ws = (char*)d_ws;                       // 288 MiB used
    u16* Qb = (u16*)(ws);                         //  8 MiB [B,8,H,W]
    u16* Kb = (u16*)(ws + 8388608);               //  8 MiB   (contiguous with Qb)
    u16* Vb = (u16*)(ws + 16777216);              // 64 MiB   (contiguous with Kb)
    u16* Qt = (u16*)(ws + 83886080);              //  8 MiB [B,8,W,H]
    u16* Kt = (u16*)(ws + 92274688);              //  8 MiB
    u16* Vt = (u16*)(ws + 100663296);             // 64 MiB [B,64,W,H]
    float* oHt = (float*)(ws + 167772160);        // 128 MiB [B,64,W,H] fp32

    proj_kernel<<<2048, 256, 0, stream>>>(x, wq, bq, wk, bk, wvw, bv, Qb, Kb, Vb);
    // Qb|Kb|Vb and Qt|Kt|Vt are each contiguous: one launch transposes all 2560 planes
    transpose_kernel<<<2560, 256, 0, stream>>>(Qb, Qt);
    // column pass: lines are W-columns, writes raw oH (transposed layout)
    attn_kernel<true><<<4096, 256, 0, stream>>>(Qt, Kt, Vt, oHt, nullptr);
    // row pass: lines are H-rows, writes gamma*oW directly into d_out
    attn_kernel<false><<<4096, 256, 0, stream>>>(Qb, Kb, Vb, out, gamma);
    combine_kernel<<<4096, 256, 0, stream>>>(oHt, x, gamma, out);
}
